// Module_7318624272490
// MI455X (gfx1250) — matrix operand order repaired and checked by reading, not compiled, not run on hardware
//
#include <hip/hip_runtime.h>
#include <hip/hip_bf16.h>
#include <stdint.h>

typedef __attribute__((ext_vector_type(16))) _Float16 v16h;
typedef __attribute__((ext_vector_type(8)))  float    v8f;
typedef __attribute__((ext_vector_type(8)))  uint32_t v8u;

#define N_USERS  20000
#define N_ITEMS  10000
#define ROWLEN   10001   // interactions row length (n_items+1)
#define NB       4096    // batch
#define NF       64      // embedding dim
#define KU_PAD   10016   // 313*32 padded K for user GEMM
#define KU_STEPS 313
#define KI_STEPS 625     // 20000/32 exactly
#define SPLITS   8

// ---------------- weight conversion f32 -> f16 ----------------
__global__ void cvt_wu(const float* __restrict__ w, _Float16* __restrict__ out) {
    int idx = blockIdx.x * 256 + threadIdx.x;
    if (idx >= NF * KU_PAD) return;
    int f = idx / KU_PAD, k = idx - f * KU_PAD;
    out[idx] = (k < N_ITEMS) ? (_Float16)w[f * N_ITEMS + k] : (_Float16)0.0f;
}

__global__ void cvt_wi(const float* __restrict__ w, _Float16* __restrict__ out) {
    int idx = blockIdx.x * 256 + threadIdx.x;
    if (idx >= NF * N_USERS) return;
    out[idx] = (_Float16)w[idx];
}

// ------------- user-path: pack masked row to u8 + fused normalizer -------------
// One block per b: read interactions[user_idx[b], 0..9999] coalesced once,
// zero the item_idx[b] column, write dense u8 row (K padded to 10016),
// and reduce the masked sum -> inv_su[b] = 1/sqrt(clip(sum,1)).
__global__ void pack_user(const int* __restrict__ inter,
                          const int* __restrict__ user_idx,
                          const int* __restrict__ item_idx,
                          uint8_t* __restrict__ Su,
                          float* __restrict__ inv_su) {
    __shared__ int red[256];
    const int b = blockIdx.x;
    const long rowoff = (long)user_idx[b] * ROWLEN;
    const int maskc = item_idx[b];
    uint8_t* out = Su + (size_t)b * KU_PAD;
    int s = 0;
    for (int k = threadIdx.x; k < KU_PAD; k += 256) {
        int v = 0;
        if (k < N_ITEMS && k != maskc) v = inter[rowoff + k];
        out[k] = (uint8_t)v;
        s += v;
    }
    red[threadIdx.x] = s; __syncthreads();
    for (int st = 128; st > 0; st >>= 1) {
        if (threadIdx.x < st) red[threadIdx.x] += red[threadIdx.x + st];
        __syncthreads();
    }
    if (threadIdx.x == 0) {
        float ss = (float)red[0];
        if (ss < 1.0f) ss = 1.0f;
        inv_su[b] = rsqrtf(ss);
    }
}

// ------------- item-path column gather via LDS row staging -------------
// One block per user-row u: stage the 40KB row in LDS (coalesced read),
// gather the 4096 sampled item columns, mask u==user_idx[b], write
// G_tmp[u][b] (coalesced byte writes).
__global__ void gather_cols(const int* __restrict__ inter,
                            const int* __restrict__ user_idx,
                            const int* __restrict__ item_idx,
                            uint8_t* __restrict__ Gt) {
    __shared__ int srow[N_ITEMS];
    const int u = blockIdx.x;
    const long rowoff = (long)u * ROWLEN;
    for (int k = threadIdx.x; k < N_ITEMS; k += 256) srow[k] = inter[rowoff + k];
    __syncthreads();
    uint8_t* out = Gt + (size_t)u * NB;
    for (int b = threadIdx.x; b < NB; b += 256) {
        int v = srow[item_idx[b]];
        if (user_idx[b] == u) v = 0;
        out[b] = (uint8_t)v;
    }
}

// ------------- transpose G_tmp[u][b] -> G[b][u] (LDS tiled) -------------
__global__ void transpose_g(const uint8_t* __restrict__ Gt, uint8_t* __restrict__ G) {
    __shared__ uint8_t tile[32][33];
    const int u0 = blockIdx.x * 32;
    const int b0 = blockIdx.y * 32;
    const int tx = threadIdx.x, ty = threadIdx.y;  // 32 x 8
    #pragma unroll
    for (int r = 0; r < 4; ++r)
        tile[ty + 8 * r][tx] = Gt[(size_t)(u0 + ty + 8 * r) * NB + b0 + tx];
    __syncthreads();
    #pragma unroll
    for (int r = 0; r < 4; ++r)
        G[(size_t)(b0 + ty + 8 * r) * N_USERS + u0 + tx] = tile[tx][ty + 8 * r];
}

// ------------- item-path sums from G[b][u] (mask pre-applied) -------------
__global__ void item_norm(const uint8_t* __restrict__ G, float* __restrict__ inv_si) {
    __shared__ int red[256];
    const int b = blockIdx.x;
    const uint8_t* row = G + (size_t)b * N_USERS;
    int s = 0;
    for (int u = threadIdx.x; u < N_USERS; u += 256) s += row[u];
    red[threadIdx.x] = s; __syncthreads();
    for (int st = 128; st > 0; st >>= 1) {
        if (threadIdx.x < st) red[threadIdx.x] += red[threadIdx.x + st];
        __syncthreads();
    }
    if (threadIdx.x == 0) {
        float ss = (float)red[0];
        if (ss < 1.0f) ss = 1.0f;
        inv_si[b] = rsqrtf(ss);
    }
}

// Convert a dword of four 0/1 bytes into two packed-f16 dwords (1.0h/0.0h).
__device__ inline uint32_t pk01_lo(uint32_t w) {
    return ((w & 0xFFu) | ((w & 0xFF00u) << 8)) * 0x3C00u;
}
__device__ inline uint32_t pk01_hi(uint32_t w) {
    return (((w >> 16) & 0xFFu) | ((w >> 24) << 16)) * 0x3C00u;
}

// ------------- unified u8 GEMM: hist[b][f] += A(b,:) @ W(:,f) -------------
// A: u8 [NB][ldA] (values 0/1, mask pre-applied, K padded/multiple of 32)
// W: f16 [NF][ldA] (B[k,n] = W[n][k]); split-K with f32 atomics.
// Contiguous k order, not the ISA table order: A-lane M=lane&15, K = k0 + (lane>=16)*16 + 0..15;
// B-lane N=lane&15, K = k0 + (lane>=16)*16 + 0..15.
__global__ void gemm_u8(const uint8_t* __restrict__ A,
                        const _Float16* __restrict__ W,
                        float* __restrict__ hist,
                        int ldA, int total_steps, int per_split) {
    const int lane = threadIdx.x & 31;
    const int wave = threadIdx.x >> 5;      // -> f-tile (0..3)
    const int btile = blockIdx.x;           // 0..255
    const int split = blockIdx.y;           // 0..SPLITS-1
    const int m  = lane & 15;
    const int hi = lane >> 4;

    const int b0 = btile * 16;
    const int f0 = wave * 16;

    const uint8_t* ap = A + (size_t)(b0 + m) * ldA;
    const _Float16* bp = W + (size_t)(f0 + m) * ldA + hi * 16;

    int ks = split * per_split;
    int ke = ks + per_split; if (ke > total_steps) ke = total_steps;

    v8f acc = {};
    for (int kk = ks; kk < ke; ++kk) {
        const int k0 = kk * 32;
        const int abase = k0 + hi * 16;
        const uint32_t* p0 = (const uint32_t*)(ap + abase);        // 4B aligned
        const uint32_t* p1 = (const uint32_t*)(ap + abase + 8);
        uint32_t w0 = p0[0], w1 = p0[1], w2 = p1[0], w3 = p1[1];
        v8u hp;
        hp[0] = pk01_lo(w0); hp[1] = pk01_hi(w0);
        hp[2] = pk01_lo(w1); hp[3] = pk01_hi(w1);
        hp[4] = pk01_lo(w2); hp[5] = pk01_hi(w2);
        hp[6] = pk01_lo(w3); hp[7] = pk01_hi(w3);
        v16h a = __builtin_bit_cast(v16h, hp);
        v16h bf = *(const v16h*)(bp + k0);   // 32B aligned, contiguous
        acc = __builtin_amdgcn_wmma_f32_16x16x32_f16(false, a, false, bf,
                                                     (short)0, acc, false, false);
    }
    #pragma unroll
    for (int v = 0; v < 8; ++v) {
        int r = b0 + v + hi * 8;
        atomicAdd(&hist[r * NF + f0 + m], acc[v]);
    }
}

// ------------- finalize: embeddings + GMF + logit -------------
__global__ void finalize_k(const int* __restrict__ user_idx,
                           const int* __restrict__ item_idx,
                           const float* __restrict__ ue,
                           const float* __restrict__ ie,
                           const float* __restrict__ hist_u,
                           const float* __restrict__ hist_i,
                           const float* __restrict__ inv_su,
                           const float* __restrict__ inv_si,
                           const float* __restrict__ lw,
                           const float* __restrict__ lb,
                           float* __restrict__ out) {
    __shared__ float red[256];
    const int tid = threadIdx.x;
    const int li = tid >> 6, f = tid & 63;
    const int b = blockIdx.x * 4 + li;
    const int uid = user_idx[b], iid = item_idx[b];
    float ru = ue[(size_t)uid * NF + f] + hist_u[b * NF + f] * inv_su[b];
    float ri = ie[(size_t)iid * NF + f] + hist_i[b * NF + f] * inv_si[b];
    red[tid] = ru * ri * lw[f];
    __syncthreads();
    for (int s = 32; s > 0; s >>= 1) {
        if ((tid & 63) < s) red[tid] += red[tid + s];
        __syncthreads();
    }
    if (f == 0) out[b] = red[tid] + lb[0];
}

extern "C" void kernel_launch(void* const* d_in, const int* in_sizes, int n_in,
                              void* d_out, int out_size, void* d_ws, size_t ws_size,
                              hipStream_t stream) {
    (void)in_sizes; (void)n_in; (void)out_size; (void)ws_size;
    const int*   user_idx = (const int*)d_in[0];
    const int*   item_idx = (const int*)d_in[1];
    const int*   inter    = (const int*)d_in[2];
    const float* ue       = (const float*)d_in[3];
    const float* ie       = (const float*)d_in[4];
    const float* pu       = (const float*)d_in[5];
    const float* pi       = (const float*)d_in[6];
    const float* lw       = (const float*)d_in[7];
    const float* lb       = (const float*)d_in[8];
    float* out = (float*)d_out;

    // workspace layout (~209 MB)
    char* ws = (char*)d_ws;
    size_t off = 0;
    _Float16* Wu = (_Float16*)(ws + off); off += (size_t)NF * KU_PAD * 2;       // 1.28 MB
    _Float16* Wi = (_Float16*)(ws + off); off += (size_t)NF * N_USERS * 2;      // 2.56 MB
    uint8_t*  Su = (uint8_t*)(ws + off);  off += (size_t)NB * KU_PAD;           // 41.0 MB
    uint8_t*  Gt = (uint8_t*)(ws + off);  off += (size_t)N_USERS * NB;          // 81.9 MB
    uint8_t*  G  = (uint8_t*)(ws + off);  off += (size_t)NB * N_USERS;          // 81.9 MB
    float* hist_u = (float*)(ws + off);   off += (size_t)NB * NF * 4;           // 1.05 MB
    float* hist_i = (float*)(ws + off);   off += (size_t)NB * NF * 4;           // 1.05 MB
    float* inv_su = (float*)(ws + off);   off += (size_t)NB * 4;
    float* inv_si = (float*)(ws + off);   off += (size_t)NB * 4;

    // weight conversion
    cvt_wu<<<(NF * KU_PAD + 255) / 256, 256, 0, stream>>>(pu, Wu);
    cvt_wi<<<(NF * N_USERS + 255) / 256, 256, 0, stream>>>(pi, Wi);

    // user path: pack masked rows to u8 + fused normalizer
    pack_user<<<NB, 256, 0, stream>>>(inter, user_idx, item_idx, Su, inv_su);

    // item path: gather columns (coalesced full-matrix pass), transpose, sums
    gather_cols<<<N_USERS, 256, 0, stream>>>(inter, user_idx, item_idx, Gt);
    transpose_g<<<dim3(N_USERS / 32, NB / 32), dim3(32, 8), 0, stream>>>(Gt, G);
    item_norm<<<NB, 256, 0, stream>>>(G, inv_si);

    // zero hist accumulators (hist_u and hist_i are adjacent)
    hipMemsetAsync(hist_u, 0, (size_t)NB * NF * 4 * 2, stream);

    // WMMA GEMMs, split-K = 8 (one unified kernel, two launches)
    gemm_u8<<<dim3(NB / 16, SPLITS), 128, 0, stream>>>(Su, Wu, hist_u,
                                                       KU_PAD, KU_STEPS, 40);
    gemm_u8<<<dim3(NB / 16, SPLITS), 128, 0, stream>>>(G, Wi, hist_i,
                                                       N_USERS, KI_STEPS, 79);

    // fuse embeddings + GMF + logit
    finalize_k<<<NB / 4, 256, 0, stream>>>(user_idx, item_idx, ue, ie,
                                           hist_u, hist_i, inv_su, inv_si, lw, lb, out);
}
